// MemoryGraphBackprop_55233279427023
// MI455X (gfx1250) — matrix operand order repaired and checked by reading, not compiled, not run on hardware
//
#include <hip/hip_runtime.h>
#include <math.h>

#define BS 2
#define T 64
#define C 64
#define D 64
#define N 1024
#define KCONN 32

typedef __attribute__((ext_vector_type(16))) __bf16 v16bf;
typedef __attribute__((ext_vector_type(8)))  float  v8f;
typedef __attribute__((ext_vector_type(4)))  int    v4i;
typedef __attribute__((address_space(1))) v4i g_v4i;   // global (AS1) int4
typedef __attribute__((address_space(3))) v4i s_v4i;   // LDS (AS3) int4

union BF16x16 { v16bf v; uint4 q[2]; unsigned short s[16]; };

// gfx1250 async global->LDS copies (ASYNCcnt-tracked), if this toolchain exposes them
#if defined(__has_builtin)
#if __has_builtin(__builtin_amdgcn_global_load_async_to_lds_b128) && \
    __has_builtin(__builtin_amdgcn_s_wait_asynccnt)
#define USE_ASYNC_LDS 1
#endif
#endif

__device__ __forceinline__ unsigned short f2bf(float f) {
    unsigned int u = __float_as_uint(f);
    u += 0x7FFFu + ((u >> 16) & 1u);          // round-to-nearest-even
    return (unsigned short)(u >> 16);
}
__device__ __forceinline__ float bf2f(unsigned short h) {
    return __uint_as_float(((unsigned int)h) << 16);
}

// ---- setup: densify sparse adjacency into bf16 [N][N], compute decay ----
__global__ void k_init_adj(const int* __restrict__ conn_idx,
                           const unsigned char* __restrict__ conn_mask,
                           const float* __restrict__ conn_w,
                           const float* __restrict__ decay_logit,
                           unsigned short* __restrict__ A,
                           float* __restrict__ decay) {
    int j = blockIdx.x * blockDim.x + threadIdx.x;
    if (j >= N) return;
    unsigned short* row = A + (size_t)j * N;
    uint4 z = {0u, 0u, 0u, 0u};
    uint4* rq = (uint4*)row;
    for (int i = 0; i < N / 8; ++i) rq[i] = z;          // zero 1024 bf16
    for (int k = 0; k < KCONN; ++k) {                   // serial per row: duplicates accumulate
        int   idx = conn_idx[j * KCONN + k];
        float w   = conn_w[j * KCONN + k] * (conn_mask[j * KCONN + k] ? 1.f : 0.f);
        row[idx]  = f2bf(bf2f(row[idx]) + w);
    }
    decay[j] = 1.f / (1.f + expf(-decay_logit[j]));
}

// ---- setup: L2-normalize cc along D ----
__global__ void k_init_cc(const float* __restrict__ cc, float* __restrict__ ccn) {
    int r = blockIdx.x * blockDim.x + threadIdx.x;      // (b,t,c) row
    if (r >= BS * T * C) return;
    const float* src = cc  + (size_t)r * D;
    float*       dst = ccn + (size_t)r * D;
    float ss = 0.f;
    for (int d = 0; d < D; ++d) { float v = src[d]; ss += v * v; }
    float inv = 1.f / fmaxf(sqrtf(ss), 1e-8f);
    for (int d = 0; d < D; ++d) dst[d] = src[d] * inv;
}

// ---- setup: h from h0, pmT (transposed bf16) from prev_msg0 ----
__global__ void k_init_state(const float* __restrict__ h0, const float* __restrict__ pm0,
                             float* __restrict__ h, unsigned short* __restrict__ pmT) {
    int i = blockIdx.x * blockDim.x + threadIdx.x;      // (b,n)
    if (i >= BS * N) return;
    int b = i / N, n = i % N;
    for (int d = 0; d < D; ++d) {
        size_t src = ((size_t)b * N + n) * D + d;
        h[src] = h0[src];
        pmT[((size_t)b * D + d) * N + n] = f2bf(pm0[src]);
    }
}

// ---- one recurrence step: dense A@pm via WMMA bf16, fused state update ----
__global__ void __launch_bounds__(256) k_step(
    const unsigned short* __restrict__ A,      // [N][N] bf16
    const unsigned short* __restrict__ pmIn,   // [BS][D][N] bf16 (transposed)
    unsigned short* __restrict__ pmOut,        // [BS][D][N] bf16
    float* __restrict__ h,                     // [BS][N][D]
    const float* __restrict__ decay,           // [N]
    const float* __restrict__ ccn,             // [BS][T][C][D]
    const float* __restrict__ prim,            // [N][D]
    const unsigned char* __restrict__ eot,     // [BS][T] (numpy bool bytes)
    float* __restrict__ out,                   // [BS][T][C][D]
    int t)
{
    __shared__ unsigned short lA[16 * N];      // 32KB A strip (rows n0..n0+15, contiguous)
    int nTile = blockIdx.x;                    // 0..63
    int n0 = nTile * 16;
#if defined(USE_ASYNC_LDS) && defined(__AMDGCN__)
    {
        // direct global->LDS async copy, no VGPR round-trip; per-lane 16B transfers
        const uint4* src = (const uint4*)(A + (size_t)n0 * N);
        uint4*       dst = (uint4*)lA;
        for (int i = threadIdx.x; i < (16 * N * 2) / 16; i += blockDim.x) {
            __builtin_amdgcn_global_load_async_to_lds_b128(
                (g_v4i*)(src + i), (s_v4i*)(dst + i), 0, 0);
        }
        __builtin_amdgcn_s_wait_asynccnt(0);
    }
#else
    {
        const uint4* src = (const uint4*)(A + (size_t)n0 * N);
        uint4*       dst = (uint4*)lA;
        for (int i = threadIdx.x; i < (16 * N * 2) / 16; i += blockDim.x) dst[i] = src[i];
    }
#endif
    __syncthreads();

    int wave  = threadIdx.x >> 5;              // 8 waves: b = wave>>2, dTile = wave&3
    int lane  = threadIdx.x & 31;
    int half  = lane >> 4;
    int mrow  = lane & 15;
    int b     = wave >> 2;
    int d0    = (wave & 3) * 16;

    v8f acc = {};
    const unsigned short* pmB  = pmIn + ((size_t)b * D + (d0 + mrow)) * N; // fixed d, contiguous in m
    const unsigned short* aRow = lA + mrow * N;

    for (int kt = 0; kt < N / 32; ++kt) {
        int kb = kt * 32;
        BF16x16 av, bv;
        // A 16x32 bf16 fragment: lane half selects K chunks {0..7,16..23} vs {8..15,24..31}
        av.q[0] = *(const uint4*)(aRow + kb + half * 8);
        av.q[1] = *(const uint4*)(aRow + kb + 16 + half * 8);
        // B 32x16 fragment: lane half selects K chunks {0..7,16..23} vs {8..15,24..31} at N = lane&15 (via pmT)
        const unsigned short* bp = pmB + kb + half * 8;
        bv.q[0] = *(const uint4*)(bp);
        bv.q[1] = *(const uint4*)(bp + 16);
        acc = __builtin_amdgcn_wmma_f32_16x16x32_bf16(false, av.v, false, bv.v,
                                                      (short)0, acc, false, false);
    }

    // epilogue: C/D layout -> VGPR r is (n = n0 + r + 8*half, d = d0 + (lane&15))
    float eo = eot[b * T + t] ? 1.f : 0.f;
    for (int r = 0; r < 8; ++r) {
        int n = n0 + r + 8 * half;
        int d = d0 + mrow;
        float recv = acc[r];
        if (n < C) recv += ccn[(((size_t)b * T + t) * C + n) * D + d];
        float dt = decay[n] * (1.f - eo);
        size_t hi = ((size_t)b * N + n) * D + d;
        float hv = dt * h[hi] + (1.f - dt) * recv;
        h[hi] = hv;
        float pm = tanhf(hv * prim[(size_t)n * D + d]);
        pmOut[((size_t)b * D + d) * N + n] = f2bf(pm);
        if (n < C) out[(((size_t)b * T + t) * C + n) * D + d] = pm;
    }
}

extern "C" void kernel_launch(void* const* d_in, const int* in_sizes, int n_in,
                              void* d_out, int out_size, void* d_ws, size_t ws_size,
                              hipStream_t stream) {
    const float*         cc     = (const float*)d_in[0];
    const unsigned char* eot    = (const unsigned char*)d_in[1];
    const int*           cidx   = (const int*)d_in[2];
    const unsigned char* cmask  = (const unsigned char*)d_in[3];
    const float*         prim   = (const float*)d_in[4];
    const float*         cw     = (const float*)d_in[5];
    const float*         dlogit = (const float*)d_in[6];
    const float*         h0     = (const float*)d_in[7];
    const float*         pm0    = (const float*)d_in[8];

    char* ws = (char*)d_ws;
    unsigned short* A    = (unsigned short*)(ws + 0);        // 2 MB  bf16 [N][N]
    float*          ccn  = (float*)(ws + 2097152);           // 2 MB  f32  normalized cc
    float*          h    = (float*)(ws + 4194304);           // 512KB f32  state
    float*          dec  = (float*)(ws + 4718592);           // 4 KB  f32  decay
    unsigned short* pmT0 = (unsigned short*)(ws + 4722688);  // 256KB bf16 pm transposed (buf0)
    unsigned short* pmT1 = (unsigned short*)(ws + 4984832);  // 256KB bf16 pm transposed (buf1)

    k_init_adj  <<<4, 256, 0, stream>>>(cidx, cmask, cw, dlogit, A, dec);
    k_init_cc   <<<32, 256, 0, stream>>>(cc, ccn);
    k_init_state<<<8, 256, 0, stream>>>(h0, pm0, h, pmT0);

    float* out = (float*)d_out;
    for (int t = 0; t < T; ++t) {
        const unsigned short* pin  = (t & 1) ? pmT1 : pmT0;
        unsigned short*       pout = (t & 1) ? pmT0 : pmT1;
        k_step<<<64, 256, 0, stream>>>(A, pin, pout, h, dec, ccn, prim, eot, out, t);
    }
}
